// Attention_50646254355255
// MI455X (gfx1250) — matrix operand order repaired and checked by reading, not compiled, not run on hardware
//
#include <hip/hip_runtime.h>
#include <stdint.h>

// ---------------------------------------------------------------------------
// Problem constants (match the JAX reference)
// ---------------------------------------------------------------------------
namespace {
constexpr int kB    = 2;
constexpr int kT    = 2048;
constexpr int kHID  = 2048;
constexpr int kH    = 32;     // query heads
constexpr int kKV   = 8;      // kv heads
constexpr int kD    = 64;     // head dim
constexpr int kWIN  = 256;    // sliding window
// ln(10000)/32 : inv_freq[d] = exp(-d * kLogThetaOver), d in [0,32)
constexpr float kLogThetaOver = 0.28782313662425575f;
}

// ---------------------------------------------------------------------------
// WMMA / TDM types (CDNA5 / gfx1250, wave32)
// ---------------------------------------------------------------------------
typedef __attribute__((ext_vector_type(16))) __bf16 v16bf;
typedef __attribute__((ext_vector_type(8)))  float  v8f;
typedef __attribute__((ext_vector_type(4)))  unsigned int v4u;
typedef __attribute__((ext_vector_type(8)))  int v8i;
typedef __attribute__((ext_vector_type(4)))  int v4i;

struct alignas(16) B16 { unsigned int w[4]; };   // 16-byte chunk (8 bf16)
struct alignas(16) B32 { B16 lo; B16 hi; };      // 32-byte fragment (16 bf16)

__device__ __forceinline__ v16bf make_frag(const unsigned short* p_lo,
                                           const unsigned short* p_hi) {
  B32 t;
  t.lo = *(const B16*)p_lo;
  t.hi = *(const B16*)p_hi;
  return __builtin_bit_cast(v16bf, t);
}

__device__ __forceinline__ unsigned short f32_to_bf16_rne(float f) {
  union { float f; unsigned int u; } a; a.f = f;
  unsigned int u = a.u;
  u += 0x7FFFu + ((u >> 16) & 1u);
  return (unsigned short)(u >> 16);
}

__device__ __forceinline__ float rowmax16(float v) {
  v = fmaxf(v, __shfl_xor(v, 1, 32));
  v = fmaxf(v, __shfl_xor(v, 2, 32));
  v = fmaxf(v, __shfl_xor(v, 4, 32));
  v = fmaxf(v, __shfl_xor(v, 8, 32));
  return v;
}
__device__ __forceinline__ float rowsum16(float v) {
  v += __shfl_xor(v, 1, 32);
  v += __shfl_xor(v, 2, 32);
  v += __shfl_xor(v, 4, 32);
  v += __shfl_xor(v, 8, 32);
  return v;
}

// Generic LDS pointer -> LDS byte offset (flat aperture keeps it in addr[31:0]).
__device__ __forceinline__ unsigned lds_byte_addr(const void* p) {
  return (unsigned)(unsigned long long)p;
}

// ---------------------------------------------------------------------------
// TDM: 2-D tensor tile (32 x 128 rows of bf16) from global [nrows][rowlen]
// into LDS, per CDNA5 D# descriptor layout (CDNA5 ISA).
// Issued by one wave; tracked with TENSORcnt.
// ---------------------------------------------------------------------------
__device__ __forceinline__ void tdm_load_tile_2d(unsigned lds_byte,
                                                 const void* gtile,
                                                 unsigned rowlen_elts,
                                                 unsigned nrows_total) {
  const unsigned long long ga = (unsigned long long)gtile;
  v4u g0;
  g0[0] = 1u;                                           // count=1 (valid), user mode
  g0[1] = lds_byte;                                     // lds_addr
  g0[2] = (unsigned)ga;                                 // global_addr[31:0]
  g0[3] = ((unsigned)(ga >> 32) & 0x01ffffffu)          // global_addr[56:32]
          | 0x80000000u;                                // type=2 ("image")
  v8i g1;
  g1[0] = 0x00010000;                                   // data_size=1 (2 bytes)
  g1[1] = (int)((rowlen_elts & 0xffffu) << 16);         // tensor_dim0[15:0]
  g1[2] = (int)(((rowlen_elts >> 16) & 0xffffu)         // tensor_dim0[31:16]
                | ((nrows_total & 0xffffu) << 16));     // tensor_dim1[15:0]
  g1[3] = (int)(((nrows_total >> 16) & 0xffffu)         // tensor_dim1[31:16]
                | (32u << 16));                         // tile_dim0 = 32
  g1[4] = 128;                                          // tile_dim1 = 128, tile_dim2 = 0
  g1[5] = (int)rowlen_elts;                             // tensor_dim0_stride[31:0]
  g1[6] = 0;                                            // stride0 hi / stride1 lo
  g1[7] = 0;                                            // stride1 hi
  v4i z4 = {0, 0, 0, 0};
#if defined(__clang_major__) && (__clang_major__ >= 23)
  v8i z8 = {0, 0, 0, 0, 0, 0, 0, 0};
  __builtin_amdgcn_tensor_load_to_lds(g0, g1, z4, z4, z8, 0);
#else
  __builtin_amdgcn_tensor_load_to_lds(g0, g1, z4, z4, 0);
#endif
}

// ---------------------------------------------------------------------------
// fp32 -> bf16 converts
// ---------------------------------------------------------------------------
__global__ void cvt_f32_bf16_v4(const float* __restrict__ src,
                                unsigned short* __restrict__ dst, int n4) {
  int i = blockIdx.x * blockDim.x + threadIdx.x;   // one float4 per thread
  if (i >= n4) return;
  const float4 f = ((const float4*)src)[i];
  union { unsigned short s[4]; uint2 u; } p;
  p.s[0] = f32_to_bf16_rne(f.x);
  p.s[1] = f32_to_bf16_rne(f.y);
  p.s[2] = f32_to_bf16_rne(f.z);
  p.s[3] = f32_to_bf16_rne(f.w);
  ((uint2*)dst)[i] = p.u;
}

// W[K][N] -> Wt[N][K] with bf16 convert (coalesced read, scattered write)
__global__ void cvtT_f32_bf16(const float* __restrict__ src,
                              unsigned short* __restrict__ dst, int K, int N) {
  int i = blockIdx.x * blockDim.x + threadIdx.x;
  if (i < K * N) {
    int k = i / N;
    int n = i - k * N;
    dst[(size_t)n * K + k] = f32_to_bf16_rne(src[i]);
  }
}

// ---------------------------------------------------------------------------
// GEMM: C[M][N] = A[M][K] * Bt[N][K]^T + bias  (bf16 in, f32 accumulate)
// 128 threads = 4 waves (2x2); block tile 128x128; wave tile 64x64 (16 WMMAs).
// Tile staging by the Tensor Data Mover (double-buffered, wave 0 issues,
// s_wait_tensorcnt + barrier to hand off); USE_TDM=false falls back to
// per-lane global->LDS staging.
// MODE 1: f32 row-major out.
// MODE 2: bf16 out transposed into V layout [B][KV][D][T].
// MODE 3: bf16 row-major out with fused RoPE (wave's 64 cols == one head).
// ---------------------------------------------------------------------------
template <int MODE, bool USE_TDM>
__global__ __launch_bounds__(128)
void gemm_wmma(const unsigned short* __restrict__ A,
               const unsigned short* __restrict__ Bt,
               const float* __restrict__ bias,
               void* __restrict__ Cout,
               int M, int N, int K) {
  __shared__ alignas(16) unsigned short Al[2][128 * 32];
  __shared__ alignas(16) unsigned short Bl[2][128 * 32];

  const int tid  = threadIdx.x;
  const int lane = tid & 31;
  const int wid  = tid >> 5;
  const int wm   = wid & 1;          // wave M position (0..1) -> 64 rows each
  const int wn   = wid >> 1;         // wave N position (0..1) -> 64 cols each
  const int half = lane >> 4;
  const int nl   = lane & 15;
  const int m0   = blockIdx.y * 128;
  const int n0   = blockIdx.x * 128;

  v8f acc[4][4];
#pragma unroll
  for (int sm = 0; sm < 4; ++sm)
#pragma unroll
    for (int sn = 0; sn < 4; ++sn)
#pragma unroll
      for (int j = 0; j < 8; ++j) acc[sm][sn][j] = 0.0f;

  // stage K-tile `k0` into buffer `buf`
  auto stage = [&](int buf, int k0) {
    if (USE_TDM) {
      if (wid == 0) {
        tdm_load_tile_2d(lds_byte_addr(&Al[buf][0]),
                         A + (size_t)m0 * K + k0, (unsigned)K, (unsigned)M);
        tdm_load_tile_2d(lds_byte_addr(&Bl[buf][0]),
                         Bt + (size_t)n0 * K + k0, (unsigned)K, (unsigned)N);
      }
    } else {
      // one thread per tile row: 64B from global, 64B to LDS
      const unsigned short* ap = A  + (size_t)(m0 + tid) * K + k0;
      const unsigned short* bp = Bt + (size_t)(n0 + tid) * K + k0;
#pragma unroll
      for (int c = 0; c < 4; ++c) {
        *(B16*)&Al[buf][tid * 32 + c * 8] = *(const B16*)(ap + c * 8);
        *(B16*)&Bl[buf][tid * 32 + c * 8] = *(const B16*)(bp + c * 8);
      }
    }
  };
  auto stage_wait = [&]() {
    if (USE_TDM) {
      if (wid == 0) __builtin_amdgcn_s_wait_tensorcnt(0);
    }
    __syncthreads();
  };

  stage(0, 0);
  stage_wait();

  int cur = 0;
  for (int k0 = 0; k0 < K; k0 += 32) {
    if (k0 + 32 < K) stage(cur ^ 1, k0 + 32);   // DMA next tile during compute

    v16bf af[4], bfr[4];
#pragma unroll
    for (int sm = 0; sm < 4; ++sm) {
      const int r = wm * 64 + sm * 16 + nl;          // A row (M)
      af[sm] = make_frag(&Al[cur][r * 32 + 8 * half],
                         &Al[cur][r * 32 + 16 + 8 * half]);
    }
#pragma unroll
    for (int sn = 0; sn < 4; ++sn) {
      const int r = wn * 64 + sn * 16 + nl;          // B column (N)
      bfr[sn] = make_frag(&Bl[cur][r * 32 + 8 * half],
                          &Bl[cur][r * 32 + 16 + 8 * half]);
    }
#pragma unroll
    for (int sm = 0; sm < 4; ++sm)
#pragma unroll
      for (int sn = 0; sn < 4; ++sn)
        acc[sm][sn] = __builtin_amdgcn_wmma_f32_16x16x32_bf16(
            false, af[sm], false, bfr[sn], (short)0, acc[sm][sn], false, false);

    stage_wait();   // next buffer ready for everyone; reads of `cur` done
    cur ^= 1;
  }

  // epilogue: C layout is (lane: col n = nl, rows m = j + 8*half)
  if (MODE == 3) {
    // fused bias + RoPE: wave's 64 columns are exactly one head; the rotation
    // partner (d, d+32) sits in acc[sm][sn] / acc[sm][sn+2] of this wave.
#pragma unroll
    for (int sm = 0; sm < 4; ++sm) {
#pragma unroll
      for (int sn = 0; sn < 2; ++sn) {
        const int d  = sn * 16 + nl;                 // 0..31 within head
        const int n1 = n0 + wn * 64 + sn * 16 + nl;
        const int n2 = n1 + 32;
        const float b1 = bias[n1];
        const float b2 = bias[n2];
        const float invf = __expf(-kLogThetaOver * (float)d);
#pragma unroll
        for (int j = 0; j < 8; ++j) {
          const int m = m0 + wm * 64 + sm * 16 + j + 8 * half;
          const int t = m & (kT - 1);
          float s, c;
          __sincosf((float)t * invf, &s, &c);
          const float v1 = acc[sm][sn][j]     + b1;
          const float v2 = acc[sm][sn + 2][j] + b2;
          ((unsigned short*)Cout)[(size_t)m * N + n1] = f32_to_bf16_rne(v1 * c - v2 * s);
          ((unsigned short*)Cout)[(size_t)m * N + n2] = f32_to_bf16_rne(v2 * c + v1 * s);
        }
      }
    }
  } else {
#pragma unroll
    for (int sm = 0; sm < 4; ++sm) {
#pragma unroll
      for (int sn = 0; sn < 4; ++sn) {
        const int n = n0 + wn * 64 + sn * 16 + nl;
        const float bn = bias[n];
#pragma unroll
        for (int j = 0; j < 8; ++j) {
          const int m = m0 + wm * 64 + sm * 16 + j + 8 * half;
          const float v = acc[sm][sn][j] + bn;
          if (MODE == 1) {
            ((float*)Cout)[(size_t)m * N + n] = v;
          } else {
            // MODE 2 — V transposed: [B][KV][D][T]
            const int kvh = n >> 6, d = n & 63;
            const int bb  = m >> 11, t = m & (kT - 1);
            ((unsigned short*)Cout)[(((size_t)bb * kKV + kvh) * kD + d) * kT + t] =
                f32_to_bf16_rne(v);
          }
        }
      }
    }
  }
}

// ---------------------------------------------------------------------------
// Sliding-window GQA attention with sink, flash-style online softmax.
// Block = 128 threads = 4 waves; each wave owns one (b, h, 16-query tile).
// ---------------------------------------------------------------------------
__global__ __launch_bounds__(128)
void attn_kernel(const unsigned short* __restrict__ qb,   // [B*T][H*D]
                 const unsigned short* __restrict__ kb,   // [B*T][KV*D]
                 const unsigned short* __restrict__ vT,   // [B][KV][D][T]
                 const float* __restrict__ sinks,         // [H]
                 unsigned short* __restrict__ ab) {       // [B*T][H*D]
  __shared__ alignas(16) unsigned short Plds[4][16 * 32];

  const int lane = threadIdx.x & 31;
  const int wid  = threadIdx.x >> 5;
  const int g    = blockIdx.x * 4 + wid;
  const int ntq  = kT / 16;
  const int qt   = g % ntq;
  const int h    = (g / ntq) % kH;
  const int b    = g / (ntq * kH);
  const int q0   = qt * 16;
  const int kvh  = h >> 2;                 // G = 4
  const int nl   = lane & 15;
  const int half = lane >> 4;

  // Q fragments for both 32-wide K chunks of D=64 (A layout: row m = nl)
  const unsigned short* qrow = qb + ((size_t)(b * kT + q0 + nl) * kH + h) * kD;
  v16bf aq[2];
#pragma unroll
  for (int kc = 0; kc < 2; ++kc)
    aq[kc] = make_frag(qrow + kc * 32 + 8 * half, qrow + kc * 32 + 16 + 8 * half);

  // online softmax state: sink folded in as initial max/sum
  const float sv = sinks[h];
  float rM[8], rS[8];
  v8f O[4];
#pragma unroll
  for (int j = 0; j < 8; ++j) { rM[j] = sv; rS[j] = 1.0f; }
#pragma unroll
  for (int dc = 0; dc < 4; ++dc)
#pragma unroll
    for (int j = 0; j < 8; ++j) O[dc][j] = 0.0f;

  int lo = q0 - (kWIN - 1);
  if (lo < 0) lo = 0;
  lo &= ~31;
  const int hi = q0 + 15;
  const size_t vhead = ((size_t)b * kKV + kvh) * kD;  // row index into vT (x kT)

  unsigned short* pl = Plds[wid];

  for (int base = lo; base <= hi; base += 32) {
    // ---- S = Q * K^T for 32 keys (two 16-key column tiles) ----
    v8f s0, s1;
#pragma unroll
    for (int j = 0; j < 8; ++j) { s0[j] = 0.0f; s1[j] = 0.0f; }
    const unsigned short* kr0 =
        kb + ((size_t)(b * kT + base + nl) * kKV + kvh) * kD;
    const unsigned short* kr1 = kr0 + (size_t)16 * kKV * kD;
#pragma unroll
    for (int kc = 0; kc < 2; ++kc) {
      v16bf bk0 = make_frag(kr0 + kc * 32 + 8 * half, kr0 + kc * 32 + 16 + 8 * half);
      v16bf bk1 = make_frag(kr1 + kc * 32 + 8 * half, kr1 + kc * 32 + 16 + 8 * half);
      s0 = __builtin_amdgcn_wmma_f32_16x16x32_bf16(false, aq[kc], false, bk0,
                                                   (short)0, s0, false, false);
      s1 = __builtin_amdgcn_wmma_f32_16x16x32_bf16(false, aq[kc], false, bk1,
                                                   (short)0, s1, false, false);
    }

    // ---- mask + scale + online softmax update (no EXEC divergence) ----
    const int k0a = base + nl;
    const int k1a = k0a + 16;
    float p0v[8], p1v[8];
#pragma unroll
    for (int j = 0; j < 8; ++j) {
      const int q = q0 + j + 8 * half;
      float x0 = s0[j] * 0.125f;   // D^-0.5
      float x1 = s1[j] * 0.125f;
      if (!((k0a <= q) && (q - k0a < kWIN))) x0 = -1e30f;
      if (!((k1a <= q) && (q - k1a < kWIN))) x1 = -1e30f;
      const float bm = rowmax16(fmaxf(x0, x1));
      const float Mn = fmaxf(rM[j], bm);
      const float al = __expf(rM[j] - Mn);
      const float e0 = __expf(x0 - Mn);
      const float e1 = __expf(x1 - Mn);
      rS[j] = rS[j] * al + rowsum16(e0 + e1);
      rM[j] = Mn;
      p0v[j] = e0;
      p1v[j] = e1;
#pragma unroll
      for (int dc = 0; dc < 4; ++dc) O[dc][j] *= al;
    }

    // ---- route P (C layout) -> A layout through LDS ----
#pragma unroll
    for (int j = 0; j < 8; ++j) {
      pl[(j + 8 * half) * 32 + nl]      = f32_to_bf16_rne(p0v[j]);
      pl[(j + 8 * half) * 32 + 16 + nl] = f32_to_bf16_rne(p1v[j]);
    }
    asm volatile("s_wait_dscnt 0" ::: "memory");  // cross-lane LDS RAW, wave-local
    v16bf pa = make_frag(&pl[nl * 32 + 8 * half], &pl[nl * 32 + 16 + 8 * half]);

    // ---- O += P * V  (V transposed in global: contiguous over keys) ----
#pragma unroll
    for (int dc = 0; dc < 4; ++dc) {
      const unsigned short* vp =
          vT + (vhead + dc * 16 + nl) * kT + base + 8 * half;
      v16bf bv = make_frag(vp, vp + 16);
      O[dc] = __builtin_amdgcn_wmma_f32_16x16x32_bf16(false, pa, false, bv,
                                                      (short)0, O[dc], false, false);
    }
  }

  // ---- normalize + store (bf16, natural [B*T][H*D] layout) ----
#pragma unroll
  for (int j = 0; j < 8; ++j) {
    const float inv = 1.0f / rS[j];
    const size_t row = ((size_t)(b * kT + q0 + j + 8 * half) * kH + h) * kD;
#pragma unroll
    for (int dc = 0; dc < 4; ++dc)
      ab[row + dc * 16 + nl] = f32_to_bf16_rne(O[dc][j] * inv);
  }
}

// ---------------------------------------------------------------------------
// Host launcher
// ---------------------------------------------------------------------------
extern "C" void kernel_launch(void* const* d_in, const int* in_sizes, int n_in,
                              void* d_out, int out_size, void* d_ws, size_t ws_size,
                              hipStream_t stream) {
  (void)in_sizes; (void)n_in; (void)out_size; (void)ws_size;

  const float* x     = (const float*)d_in[0];
  const float* Wq    = (const float*)d_in[1];
  const float* bq    = (const float*)d_in[2];
  const float* Wk    = (const float*)d_in[3];
  const float* bk    = (const float*)d_in[4];
  const float* Wv    = (const float*)d_in[5];
  const float* bv    = (const float*)d_in[6];
  const float* Wo    = (const float*)d_in[7];
  const float* bo    = (const float*)d_in[8];
  const float* sinks = (const float*)d_in[9];
  float* out = (float*)d_out;

  const int M   = kB * kT;        // 4096 tokens
  const int Nq  = kH * kD;        // 2048
  const int Nkv = kKV * kD;       // 512
  const int K   = kHID;           // 2048

  char* ws = (char*)d_ws;
  auto take = [&](size_t bytes) {
    char* p = ws;
    ws += (bytes + 255) & ~(size_t)255;
    return p;
  };
  unsigned short* xb   = (unsigned short*)take((size_t)M * K   * 2);  // 16 MB
  unsigned short* Wqt  = (unsigned short*)take((size_t)Nq * K  * 2);  //  8 MB
  unsigned short* Wkt  = (unsigned short*)take((size_t)Nkv * K * 2);  //  2 MB
  unsigned short* Wvt  = (unsigned short*)take((size_t)Nkv * K * 2);  //  2 MB
  unsigned short* Wot  = (unsigned short*)take((size_t)kHID * Nq * 2);//  8 MB
  unsigned short* qbuf = (unsigned short*)take((size_t)M * Nq  * 2);  // 16 MB
  unsigned short* kbuf = (unsigned short*)take((size_t)M * Nkv * 2);  //  4 MB
  unsigned short* vTb  = (unsigned short*)take((size_t)M * Nkv * 2);  //  4 MB
  unsigned short* abuf = (unsigned short*)take((size_t)M * Nq  * 2);  // 16 MB
  // total ~76 MB of workspace

  // 1) bf16 conversion / weight transposition
  cvt_f32_bf16_v4<<<(M * K / 4) / 256, 256, 0, stream>>>(x, xb, M * K / 4);
  cvtT_f32_bf16<<<(K * Nq)  / 256, 256, 0, stream>>>(Wq, Wqt, K, Nq);
  cvtT_f32_bf16<<<(K * Nkv) / 256, 256, 0, stream>>>(Wk, Wkt, K, Nkv);
  cvtT_f32_bf16<<<(K * Nkv) / 256, 256, 0, stream>>>(Wv, Wvt, K, Nkv);
  cvtT_f32_bf16<<<(Nq * kHID) / 256, 256, 0, stream>>>(Wo, Wot, Nq, kHID);

  // 2) QKV projections (WMMA bf16 + TDM staging); RoPE fused into Q/K epilogue
  gemm_wmma<3, true><<<dim3(Nq / 128,  M / 128), 128, 0, stream>>>(xb, Wqt, bq, qbuf, M, Nq,  K);
  gemm_wmma<3, true><<<dim3(Nkv / 128, M / 128), 128, 0, stream>>>(xb, Wkt, bk, kbuf, M, Nkv, K);
  gemm_wmma<2, true><<<dim3(Nkv / 128, M / 128), 128, 0, stream>>>(xb, Wvt, bv, vTb,  M, Nkv, K);

  // 3) windowed attention with sink (WMMA, online softmax)
  attn_kernel<<<(kB * kH * (kT / 16)) / 4, 128, 0, stream>>>(qbuf, kbuf, vTb, sinks, abuf);

  // 4) output projection -> fp32 d_out
  gemm_wmma<1, true><<<dim3(kHID / 128, M / 128), 128, 0, stream>>>(abuf, Wot, bo, out, M, kHID, Nq);
}
